// ProposalModule_56916906607088
// MI455X (gfx1250) — matrix operand order repaired and checked by reading, not compiled, not run on hardware
//
#include <hip/hip_runtime.h>
#include <hip/hip_bf16.h>

typedef __attribute__((ext_vector_type(16))) _Float16 v16h;
typedef __attribute__((ext_vector_type(8)))  _Float16 v8h;
typedef __attribute__((ext_vector_type(8)))  float    v8f;

#define Bn    32
#define Kpts  1024
#define Cn    256
#define Pn    256
#define NSn   16
#define NBP   (Bn*Pn)      /* 8192 */
#define K0PAD 288          /* 259 -> next multiple of 32 */
#define RAD   0.3f
#define RAD2  0.09f

/* output sub-array offsets (floats), concatenated in reference return order */
#define O0 0                 /* obj            (NBP,2)  */
#define O1 (O0 + NBP*2)      /* center         (NBP,3)  */
#define O2 (O1 + NBP*3)      /* heading_scores (NBP,12) */
#define O3 (O2 + NBP*12)     /* heading_res_n  (NBP,12) */
#define O4 (O3 + NBP*12)     /* heading_res    (NBP,12) */
#define O5 (O4 + NBP*12)     /* size_scores    (NBP,10) */
#define O6 (O5 + NBP*10)     /* size_res_n     (NBP,30) */
#define O7 (O6 + NBP*30)     /* size_res       (NBP,30) */
#define O8 (O7 + NBP*30)     /* sem_cls        (NBP,10) */

// ---------------------------------------------------------------------------
// WMMA fragment helpers (layouts per CDNA5 ISA §7.12.2)
// A 16-bit 16x32: lane row = lane&15, grp = lane>>4;
//   halfs e=0..7  -> K = kc + grp*8 + e          (contiguous 16B)
//   halfs e=8..15 -> K = kc + 16 + grp*8 + (e-8) (contiguous 16B)
// B 16-bit 32x16 (weights stored N-major: row n holds K contiguous):
//   lane col = lane&15; halfs e=0..15 -> same K mapping as A (two 16B chunks)
// C/D f32 16x16: lane col = lane&15; VGPR v -> row = v + 8*(lane>>4)
// ---------------------------------------------------------------------------
__device__ __forceinline__ v16h load_a_frag(const _Float16* base, int ld,
                                            int row, int kc, int grp) {
  const _Float16* p = base + row * ld + kc + grp * 8;
  v8h lo = *(const v8h*)(p);
  v8h hi = *(const v8h*)(p + 16);
  return __builtin_shufflevector(lo, hi, 0,1,2,3,4,5,6,7,8,9,10,11,12,13,14,15);
}
__device__ __forceinline__ v16h load_b_frag(const _Float16* bt, int ld,
                                            int n, int kc, int grp) {
  const _Float16* p = bt + n * ld + kc + grp * 8;
  v8h lo = *(const v8h*)(p);
  v8h hi = *(const v8h*)(p + 16);
  return __builtin_shufflevector(lo, hi, 0,1,2,3,4,5,6,7,8,9,10,11,12,13,14,15);
}
__device__ __forceinline__ v8f wmma16(v16h a, v16h b, v8f c) {
  return __builtin_amdgcn_wmma_f32_16x16x32_f16(false, a, false, b,
                                                (short)0, c, false, false);
}

// ---------------------------------------------------------------------------
// 0) Weight prep: convert to f16, transpose to N-major, zero-pad K / N
// ---------------------------------------------------------------------------
__global__ void __launch_bounds__(256)
prep_weights(const float* w0, const float* w1, const float* w2,
             const float* c1, const float* c2, const float* c3,
             _Float16* w0t, _Float16* w1t, _Float16* w2t,
             _Float16* c1t, _Float16* c2t, _Float16* c3t) {
  int i = blockIdx.x * blockDim.x + threadIdx.x;
  if (i < 128 * K0PAD) {                       // w0t[n][k], k padded 259->288
    int n = i / K0PAD, k = i - n * K0PAD;
    w0t[i] = (k < 259) ? (_Float16)w0[k * 128 + n] : (_Float16)0.0f;
    return;
  }
  i -= 128 * K0PAD;
  if (i < 16384) { int n = i >> 7, k = i & 127; w1t[i] = (_Float16)w1[k * 128 + n]; return; }
  i -= 16384;
  if (i < 16384) { int n = i >> 7, k = i & 127; w2t[i] = (_Float16)w2[k * 128 + n]; return; }
  i -= 16384;
  if (i < 16384) { int n = i >> 7, k = i & 127; c1t[i] = (_Float16)c1[k * 128 + n]; return; }
  i -= 16384;
  if (i < 16384) { int n = i >> 7, k = i & 127; c2t[i] = (_Float16)c2[k * 128 + n]; return; }
  i -= 16384;
  if (i < 80 * 128) {                          // c3t: 79 outs padded to 80 rows
    int n = i >> 7, k = i & 127;
    c3t[i] = (n < 79) ? (_Float16)c3[k * 79 + n] : (_Float16)0.0f;
  }
}

// ---------------------------------------------------------------------------
// 1) Farthest point sampling: one block per batch, one thread per point.
//    256 dependent iterations; block-wide argmax (shfl butterfly + LDS).
// ---------------------------------------------------------------------------
__global__ void __launch_bounds__(1024)
fps_kernel(const float* __restrict__ xyz, float* __restrict__ new_xyz) {
  int b = blockIdx.x, tid = threadIdx.x;
  int lane = tid & 31, wid = tid >> 5;
  __shared__ float s_val[32];
  __shared__ int   s_idx[32];
  __shared__ float bpt[3];
  __shared__ int   s_last;
  const float* base = xyz + (size_t)b * Kpts * 3;
  float x = base[tid * 3 + 0], y = base[tid * 3 + 1], z = base[tid * 3 + 2];
  float dist = 1e10f;
  int last = 0;
  for (int it = 0; it < Pn; ++it) {
    if (tid == last) { bpt[0] = x; bpt[1] = y; bpt[2] = z; }
    __syncthreads();
    float dx = x - bpt[0], dy = y - bpt[1], dz = z - bpt[2];
    dist = fminf(dist, dx * dx + dy * dy + dz * dz);
    if (tid == 0) {                   // inds[it] = last -> record its coords
      float* o = new_xyz + ((size_t)b * Pn + it) * 3;
      o[0] = bpt[0]; o[1] = bpt[1]; o[2] = bpt[2];
    }
    float v = dist; int i = tid;
    #pragma unroll
    for (int off = 16; off; off >>= 1) {       // argmax, lowest index on tie
      float vo = __shfl_xor(v, off, 32);
      int   io = __shfl_xor(i, off, 32);
      if (vo > v || (vo == v && io < i)) { v = vo; i = io; }
    }
    if (lane == 0) { s_val[wid] = v; s_idx[wid] = i; }
    __syncthreads();
    if (wid == 0) {
      v = s_val[lane]; i = s_idx[lane];
      #pragma unroll
      for (int off = 16; off; off >>= 1) {
        float vo = __shfl_xor(v, off, 32);
        int   io = __shfl_xor(i, off, 32);
        if (vo > v || (vo == v && io < i)) { v = vo; i = io; }
      }
      if (lane == 0) s_last = i;
    }
    __syncthreads();
    last = s_last;
  }
}

// ---------------------------------------------------------------------------
// 2) Ball query: one block per batch; xyz[b] staged in LDS; one thread per
//    center collects the first NS in-radius indices in ascending k (matches
//    sort-then-take-NS of the reference), padded with the first hit.
// ---------------------------------------------------------------------------
__global__ void __launch_bounds__(256)
ball_query_kernel(const float* __restrict__ xyz,
                  const float* __restrict__ new_xyz,
                  int* __restrict__ idxbuf) {
  int b = blockIdx.x;
  __shared__ float sx[Kpts], sy[Kpts], sz[Kpts];
  for (int t = threadIdx.x; t < Kpts; t += 256) {
    const float* p = xyz + ((size_t)b * Kpts + t) * 3;
    sx[t] = p[0]; sy[t] = p[1]; sz[t] = p[2];
  }
  __syncthreads();
  int p = threadIdx.x;
  const float* c = new_xyz + ((size_t)b * Pn + p) * 3;
  float cx = c[0], cy = c[1], cz = c[2];
  int out[NSn]; int cnt = 0;
  for (int k = 0; k < Kpts; ++k) {
    float dx = sx[k] - cx, dy = sy[k] - cy, dz = sz[k] - cz;
    if (dx * dx + dy * dy + dz * dz < RAD2) {
      out[cnt++] = k;
      if (cnt == NSn) break;
    }
  }
  if (cnt == 0) { out[0] = 0; cnt = 1; }       // cannot happen (center in set)
  for (int j = cnt; j < NSn; ++j) out[j] = out[0];
  int* dst = idxbuf + ((size_t)b * Pn + p) * NSn;
  for (int j = 0; j < NSn; ++j) dst[j] = out[j];
}

// ---------------------------------------------------------------------------
// 3) Fused grouped MLP: one wave per (b,p) group (NS=16 rows == one WMMA
//    M-tile). g(16x288) -> L0 -> h0(16x128) -> L1 -> h1 -> L2 -> relu ->
//    max over the 16 rows (fused into the D-fragment) -> feat[b*P+p][128].
//    All activations live in LDS; weights stream from L2.
// ---------------------------------------------------------------------------
#define MLP_WAVES 4
__global__ void __launch_bounds__(MLP_WAVES * 32)
mlp_fused(const float* __restrict__ xyz, const float* __restrict__ features,
          const int* __restrict__ idxbuf, const float* __restrict__ new_xyz,
          const _Float16* __restrict__ w0t, const float* __restrict__ s0, const float* __restrict__ b0,
          const _Float16* __restrict__ w1t, const float* __restrict__ s1, const float* __restrict__ b1,
          const _Float16* __restrict__ w2t, const float* __restrict__ s2, const float* __restrict__ b2,
          _Float16* __restrict__ feat) {
  __shared__ _Float16 g_sh[MLP_WAVES][16 * K0PAD];   // also reused as h1
  __shared__ _Float16 h0_sh[MLP_WAVES][16 * 128];
  int w = threadIdx.x >> 5, lane = threadIdx.x & 31;
  int gi = blockIdx.x * MLP_WAVES + w;               // (b,p) group id
  int bidx = gi >> 8, p = gi & 255;
  int row16 = lane & 15, grp = lane >> 4;
  _Float16* g  = g_sh[w];
  _Float16* h0 = h0_sh[w];

  // ---- gather g = [ (xyz-center)/R | features ] as f16, zero-padded K ----
  const int* myidx = idxbuf + (size_t)gi * NSn;
  float cx = new_xyz[gi * 3 + 0], cy = new_xyz[gi * 3 + 1], cz = new_xyz[gi * 3 + 2];
  for (int t = lane; t < 16 * K0PAD; t += 32) {
    int s = t / K0PAD, c = t - s * K0PAD;
    int k = myidx[s];
    float val;
    if (c < 3) {
      float pv = xyz[((size_t)bidx * Kpts + k) * 3 + c];
      float cc = (c == 0) ? cx : ((c == 1) ? cy : cz);
      val = (pv - cc) * (1.0f / RAD);
    } else if (c < 259) {
      val = features[((size_t)bidx * Kpts + k) * Cn + (c - 3)];
    } else {
      val = 0.0f;
    }
    g[t] = (_Float16)val;
  }

  // ---- layer 0: 16x288 @ 288x128 ----
  for (int nt = 0; nt < 8; ++nt) {
    v8f acc = {};
    for (int kc = 0; kc < K0PAD; kc += 32)
      acc = wmma16(load_a_frag(g, K0PAD, row16, kc, grp),
                   load_b_frag(w0t, K0PAD, nt * 16 + row16, kc, grp), acc);
    int n = nt * 16 + row16;
    float sc = s0[n], bi = b0[n];
    #pragma unroll
    for (int v = 0; v < 8; ++v) {
      float x = acc[v] * sc + bi;
      h0[(v + 8 * grp) * 128 + n] = (_Float16)(x > 0.0f ? x : 0.0f);
    }
  }

  // ---- layer 1: 16x128 @ 128x128 (h1 overlays g region) ----
  _Float16* h1 = g;
  for (int nt = 0; nt < 8; ++nt) {
    v8f acc = {};
    for (int kc = 0; kc < 128; kc += 32)
      acc = wmma16(load_a_frag(h0, 128, row16, kc, grp),
                   load_b_frag(w1t, 128, nt * 16 + row16, kc, grp), acc);
    int n = nt * 16 + row16;
    float sc = s1[n], bi = b1[n];
    #pragma unroll
    for (int v = 0; v < 8; ++v) {
      float x = acc[v] * sc + bi;
      h1[(v + 8 * grp) * 128 + n] = (_Float16)(x > 0.0f ? x : 0.0f);
    }
  }

  // ---- layer 2 + fused max over the 16 samples ----
  for (int nt = 0; nt < 8; ++nt) {
    v8f acc = {};
    for (int kc = 0; kc < 128; kc += 32)
      acc = wmma16(load_a_frag(h1, 128, row16, kc, grp),
                   load_b_frag(w2t, 128, nt * 16 + row16, kc, grp), acc);
    int n = nt * 16 + row16;
    float sc = s2[n], bi = b2[n];
    float mx = 0.0f;                              // relu output lower bound
    #pragma unroll
    for (int v = 0; v < 8; ++v) {
      float x = acc[v] * sc + bi;
      mx = fmaxf(mx, x > 0.0f ? x : 0.0f);        // 8 rows in this lane
    }
    mx = fmaxf(mx, __shfl_xor(mx, 16, 32));       // other 8 rows (lane^16)
    if (grp == 0) feat[(size_t)gi * 128 + n] = (_Float16)mx;
  }
}

// ---------------------------------------------------------------------------
// 4) Conv head: one wave per 16 proposals. feat(16x128) -> clamp6 conv1 ->
//    clamp6 conv2 -> conv3 (79 outs, N padded to 80) with the output
//    decomposition done directly from the D fragments.
// ---------------------------------------------------------------------------
#define HEAD_WAVES 4
__global__ void __launch_bounds__(HEAD_WAVES * 32)
conv_head(const _Float16* __restrict__ feat,
          const _Float16* __restrict__ c1t, const float* __restrict__ c1s, const float* __restrict__ c1b,
          const _Float16* __restrict__ c2t, const float* __restrict__ c2s, const float* __restrict__ c2b,
          const _Float16* __restrict__ c3t, const float* __restrict__ c3b,
          const float* __restrict__ mean_size, const float* __restrict__ new_xyz,
          float* __restrict__ out) {
  __shared__ _Float16 h1_sh[HEAD_WAVES][16 * 128];
  __shared__ _Float16 h2_sh[HEAD_WAVES][16 * 128];
  int w = threadIdx.x >> 5, lane = threadIdx.x & 31;
  int mt = blockIdx.x * HEAD_WAVES + w;            // 16-row tile id, < 512
  int row16 = lane & 15, grp = lane >> 4;
  _Float16* h1 = h1_sh[w];
  _Float16* h2 = h2_sh[w];

  for (int nt = 0; nt < 8; ++nt) {                 // conv1
    v8f acc = {};
    for (int kc = 0; kc < 128; kc += 32)
      acc = wmma16(load_a_frag(feat, 128, mt * 16 + row16, kc, grp),
                   load_b_frag(c1t, 128, nt * 16 + row16, kc, grp), acc);
    int n = nt * 16 + row16;
    float sc = c1s[n], bi = c1b[n];
    #pragma unroll
    for (int v = 0; v < 8; ++v) {
      float x = fminf(fmaxf(acc[v] * sc + bi, 0.0f), 6.0f);
      h1[(v + 8 * grp) * 128 + n] = (_Float16)x;
    }
  }
  for (int nt = 0; nt < 8; ++nt) {                 // conv2
    v8f acc = {};
    for (int kc = 0; kc < 128; kc += 32)
      acc = wmma16(load_a_frag(h1, 128, row16, kc, grp),
                   load_b_frag(c2t, 128, nt * 16 + row16, kc, grp), acc);
    int n = nt * 16 + row16;
    float sc = c2s[n], bi = c2b[n];
    #pragma unroll
    for (int v = 0; v < 8; ++v) {
      float x = fminf(fmaxf(acc[v] * sc + bi, 0.0f), 6.0f);
      h2[(v + 8 * grp) * 128 + n] = (_Float16)x;
    }
  }
  const float PI_NH = 3.14159265359f / 12.0f;
  for (int nt = 0; nt < 5; ++nt) {                 // conv3, N = 80 (79 valid)
    v8f acc = {};
    for (int kc = 0; kc < 128; kc += 32)
      acc = wmma16(load_a_frag(h2, 128, row16, kc, grp),
                   load_b_frag(c3t, 128, nt * 16 + row16, kc, grp), acc);
    int ch = nt * 16 + row16;
    if (ch < 79) {                                  // epilogue only diverges
      float bi = c3b[ch];
      #pragma unroll
      for (int v = 0; v < 8; ++v) {
        int m = mt * 16 + v + 8 * grp;              // proposal index in [0,8192)
        float val = acc[v] + bi;
        if (ch < 2) {
          out[O0 + m * 2 + ch] = val;                               // obj
        } else if (ch < 5) {
          int c = ch - 2;
          out[O1 + m * 3 + c] = val + new_xyz[m * 3 + c];           // center
        } else {
          int nc = ch - 3;                          // index into 76-ch "net"
          if (nc < 14) {
            out[O2 + m * 12 + (nc - 2)] = val;                      // heading_scores
          } else if (nc < 24) {
            out[O5 + m * 10 + (nc - 14)] = val;                     // size_scores
          } else if (nc < 36) {
            int j = nc - 24;
            out[O3 + m * 12 + j] = val;                             // heading_res_n
            out[O4 + m * 12 + j] = val * PI_NH;                     // heading_res
          } else if (nc < 66) {
            int j = nc - 36;
            out[O6 + m * 30 + j] = val;                             // size_res_n
            out[O7 + m * 30 + j] = val * mean_size[j];              // size_res
          } else {
            out[O8 + m * 10 + (nc - 66)] = val;                     // sem_cls
          }
        }
      }
    }
  }
}

// ---------------------------------------------------------------------------
extern "C" void kernel_launch(void* const* d_in, const int* in_sizes, int n_in,
                              void* d_out, int out_size, void* d_ws, size_t ws_size,
                              hipStream_t stream) {
  const float* xyz   = (const float*)d_in[0];
  const float* featf = (const float*)d_in[1];
  const float* msz   = (const float*)d_in[2];
  const float* w0 = (const float*)d_in[3],  *s0 = (const float*)d_in[4],  *b0 = (const float*)d_in[5];
  const float* w1 = (const float*)d_in[6],  *s1 = (const float*)d_in[7],  *b1 = (const float*)d_in[8];
  const float* w2 = (const float*)d_in[9],  *s2 = (const float*)d_in[10], *b2 = (const float*)d_in[11];
  const float* c1 = (const float*)d_in[12], *c1s = (const float*)d_in[13], *c1b = (const float*)d_in[14];
  const float* c2 = (const float*)d_in[15], *c2s = (const float*)d_in[16], *c2b = (const float*)d_in[17];
  const float* c3 = (const float*)d_in[18], *c3b = (const float*)d_in[19];
  float* out = (float*)d_out;

  // workspace carve-out (~3 MB total)
  char* ws = (char*)d_ws;
  size_t off = 0;
  auto take = [&](size_t bytes) -> void* {
    void* p = ws + off;
    off += (bytes + 255) & ~(size_t)255;
    return p;
  };
  float*     new_xyz = (float*)    take((size_t)NBP * 3 * sizeof(float));
  int*       idxb    = (int*)      take((size_t)NBP * NSn * sizeof(int));
  _Float16*  feat    = (_Float16*) take((size_t)NBP * 128 * sizeof(_Float16));
  _Float16*  w0t     = (_Float16*) take((size_t)128 * K0PAD * sizeof(_Float16));
  _Float16*  w1t     = (_Float16*) take((size_t)128 * 128 * sizeof(_Float16));
  _Float16*  w2t     = (_Float16*) take((size_t)128 * 128 * sizeof(_Float16));
  _Float16*  c1t     = (_Float16*) take((size_t)128 * 128 * sizeof(_Float16));
  _Float16*  c2t     = (_Float16*) take((size_t)128 * 128 * sizeof(_Float16));
  _Float16*  c3t     = (_Float16*) take((size_t)80 * 128 * sizeof(_Float16));
  (void)ws_size; (void)in_sizes; (void)n_in; (void)out_size;

  int prep_tasks = 128 * K0PAD + 4 * 16384 + 80 * 128;
  prep_weights<<<(prep_tasks + 255) / 256, 256, 0, stream>>>(
      w0, w1, w2, c1, c2, c3, w0t, w1t, w2t, c1t, c2t, c3t);

  fps_kernel<<<Bn, Kpts, 0, stream>>>(xyz, new_xyz);

  ball_query_kernel<<<Bn, Pn, 0, stream>>>(xyz, new_xyz, idxb);

  mlp_fused<<<NBP / MLP_WAVES, MLP_WAVES * 32, 0, stream>>>(
      xyz, featf, idxb, new_xyz,
      w0t, s0, b0, w1t, s1, b1, w2t, s2, b2, feat);

  conv_head<<<(NBP / 16) / HEAD_WAVES, HEAD_WAVES * 32, 0, stream>>>(
      feat, c1t, c1s, c1b, c2t, c2s, c2b, c3t, c3b, msz, new_xyz, out);
}
